// BiLSTMM_66563403153515
// MI455X (gfx1250) — matrix operand order repaired and checked by reading, not compiled, not run on hardware
//
#include <hip/hip_runtime.h>
#include <hip/hip_bf16.h>

// ---------------------------------------------------------------------------
// BiLSTM (T=512, B=64, D=H=512), 2 layers, bidirectional.
// bf16 WMMA (v_wmma_f32_16x16x32_bf16) for all GEMMs, fused input projection,
// persistent per-layer kernel, one device barrier per step, LSTM pointwise on
// the WMMA accumulator registers. DIN is a template constant so all inner-loop
// addressing folds to immediate offsets.
// ---------------------------------------------------------------------------

typedef __attribute__((ext_vector_type(16))) __bf16 v16bf;
typedef __attribute__((ext_vector_type(8)))  __bf16 v8bf;
typedef __attribute__((ext_vector_type(8)))  float  v8f;

#define TT 512
#define BB 64
#define HH 512
#define NBLK 16        // workgroups per direction
#define NTHREADS 256   // 8 waves per workgroup -> 128 waves per direction

// float -> bf16, round-to-nearest-even
__device__ __forceinline__ unsigned short f2bf(float f) {
  union { float f; unsigned u; } v; v.f = f;
  unsigned r = v.u + 0x7FFFu + ((v.u >> 16) & 1u);
  return (unsigned short)(r >> 16);
}

__device__ __forceinline__ float sigf(float x) {
  return 1.0f / (1.0f + __expf(-x));
}

// Assemble the 16-element bf16 operand from two contiguous 8-element chunks.
// A operand: base already includes row*ld + half*8; chunks at +k and +k+16.
__device__ __forceinline__ v16bf load_a16(const unsigned short* p) {
  v8bf lo = *(const v8bf*)(p);
  v8bf hi = *(const v8bf*)(p + 16);
  v16bf a;
#pragma unroll
  for (int i = 0; i < 8; ++i) { a[i] = lo[i]; a[8 + i] = hi[i]; }
  return a;
}
// B operand: base already includes row*ld + half*8; chunks at +0 and +16.
__device__ __forceinline__ v16bf load_b16(const unsigned short* p) {
  v8bf lo = *(const v8bf*)(p);
  v8bf hi = *(const v8bf*)(p + 16);
  v16bf b;
#pragma unroll
  for (int i = 0; i < 8; ++i) { b[i] = lo[i]; b[8 + i] = hi[i]; }
  return b;
}

// ---------------------------------------------------------------------------
__global__ void cvt_f32_to_bf16(const float* __restrict__ src,
                                unsigned short* __restrict__ dst, int n) {
  int i = blockIdx.x * blockDim.x + threadIdx.x;
  int stride = gridDim.x * blockDim.x;
  for (; i < n; i += stride) dst[i] = f2bf(src[i]);
}

__global__ void zero_u32(unsigned* __restrict__ p, int n) {
  int i = blockIdx.x * blockDim.x + threadIdx.x;
  int stride = gridDim.x * blockDim.x;
  for (; i < n; i += stride) p[i] = 0u;
}

// device-wide barrier among the NBLK blocks of one direction
__device__ __forceinline__ void grid_barrier(unsigned* cnt, unsigned* gen) {
  __syncthreads();
  __threadfence();
  if (threadIdx.x == 0) {
    unsigned g = __hip_atomic_load(gen, __ATOMIC_RELAXED, __HIP_MEMORY_SCOPE_AGENT);
    unsigned prev = __hip_atomic_fetch_add(cnt, 1u, __ATOMIC_ACQ_REL, __HIP_MEMORY_SCOPE_AGENT);
    if (prev == (unsigned)(NBLK - 1)) {
      __hip_atomic_store(cnt, 0u, __ATOMIC_RELAXED, __HIP_MEMORY_SCOPE_AGENT);
      __hip_atomic_fetch_add(gen, 1u, __ATOMIC_ACQ_REL, __HIP_MEMORY_SCOPE_AGENT);
    } else {
      while (__hip_atomic_load(gen, __ATOMIC_ACQUIRE, __HIP_MEMORY_SCOPE_AGENT) == g) {
        __builtin_amdgcn_s_sleep(2);
      }
    }
  }
  __syncthreads();
}

// ---------------------------------------------------------------------------
// One bidirectional LSTM layer. gridDim = (NBLK, 2): blockIdx.y = direction.
// xin: (T, B, DIN) bf16. W*ih: (2048, DIN), W*hh: (2048, HH) bf16 row-major.
// bias: (2048,) f32.
// ybf : optional (T, B, 1024) bf16 output (layer-0 -> layer-1 input)
// yf32: optional (T, B, 1024) f32 output (layer-1 -> d_out)
// hstate: [2 dirs][2 bufs][B*H] bf16 (double-buffered hidden state)
// cstate: [2 dirs][B*H] f32
// bar:    [2 dirs][32] u32 (cnt, gen)
// ---------------------------------------------------------------------------
template <int DIN>
__global__ void lstm_layer_kernel(
    const unsigned short* __restrict__ xin,
    const unsigned short* __restrict__ Wih_f, const unsigned short* __restrict__ Whh_f,
    const float* __restrict__ bias_f,
    const unsigned short* __restrict__ Wih_b, const unsigned short* __restrict__ Whh_b,
    const float* __restrict__ bias_b,
    const int* __restrict__ lengths,
    unsigned short* __restrict__ ybf,
    float* __restrict__ yf32,
    unsigned short* __restrict__ hstate,
    float* __restrict__ cstate,
    unsigned* __restrict__ bar)
{
  const int dir = blockIdx.y;
  const bool rev = (dir == 1);
  const unsigned short* Wih = rev ? Wih_b : Wih_f;
  const unsigned short* Whh = rev ? Whh_b : Whh_f;
  const float* bias = rev ? bias_b : bias_f;
  unsigned short* hbuf = hstate + (size_t)dir * 2 * BB * HH;
  float* cst = cstate + (size_t)dir * BB * HH;
  unsigned* cnt = bar + dir * 32;
  unsigned* gen = cnt + 1;

  const int lane = threadIdx.x & 31;
  const int wid  = (blockIdx.x * NTHREADS + (int)threadIdx.x) >> 5; // 0..127
  const int mt = wid & 3;        // batch tile (64/16 = 4)
  const int jt = wid >> 2;       // hidden-column tile (512/16 = 32)
  const int m0 = mt * 16;
  const int j0 = jt * 16;
  const int l15 = lane & 15;
  const int half = lane >> 4;
  const int jcol = j0 + l15;     // column (per-lane) within H

  // bias per gate for this lane's column
  const float bi = bias[0 * HH + jcol];
  const float bf = bias[1 * HH + jcol];
  const float bg = bias[2 * HH + jcol];
  const float bo = bias[3 * HH + jcol];

  // per-r batch row, its length, and the c-state index (fixed for all steps)
  int brow_[8], idxr_[8], lenr_[8];
#pragma unroll
  for (int r = 0; r < 8; ++r) {
    int b = m0 + r + half * 8;
    brow_[r] = b;
    idxr_[r] = b * HH + jcol;
    lenr_[r] = lengths[b];
  }

  const int am = m0 + l15;  // A-operand row (ISA A layout: M = lane%16)

  // Loop-invariant operand bases (half* folded in; compile-time strides).
  const size_t a_x_off = (size_t)am * DIN + half * 8;
  const unsigned short* ah_base0 = hbuf + (size_t)am * HH + half * 8;  // + buf sel
  const unsigned short* bih[4];
  const unsigned short* bhh[4];
#pragma unroll
  for (int g = 0; g < 4; ++g) {
    bih[g] = Wih + (size_t)(g * HH + jcol) * DIN + half * 8;
    bhh[g] = Whh + (size_t)(g * HH + jcol) * HH  + half * 8;
  }

  for (int s = 0; s < TT; ++s) {
    const int t = rev ? (TT - 1 - s) : s;
    const unsigned short* ax = xin + (size_t)t * BB * DIN + a_x_off;
    const unsigned short* ah = ah_base0 + (size_t)(s & 1) * (BB * HH);
    unsigned short* hwrite   = hbuf + (size_t)((s + 1) & 1) * (BB * HH);

    v8f acc[4];
#pragma unroll
    for (int g = 0; g < 4; ++g)
#pragma unroll
      for (int i = 0; i < 8; ++i) acc[g][i] = 0.0f;

    // gates += x_t @ Wih^T  (K = DIN, tiled by 32, immediate-offset loads)
#pragma unroll 4
    for (int kk = 0; kk < DIN; kk += 32) {
      v16bf a = load_a16(ax + kk);
#pragma unroll
      for (int g = 0; g < 4; ++g) {
        v16bf bv = load_b16(bih[g] + kk);
        acc[g] = __builtin_amdgcn_wmma_f32_16x16x32_bf16(
            false, a, false, bv, (short)0, acc[g], false, false);
      }
    }
    // gates += h @ Whh^T  (K = HH)
#pragma unroll 4
    for (int kk = 0; kk < HH; kk += 32) {
      v16bf a = load_a16(ah + kk);
#pragma unroll
      for (int g = 0; g < 4; ++g) {
        v16bf bv = load_b16(bhh[g] + kk);
        acc[g] = __builtin_amdgcn_wmma_f32_16x16x32_bf16(
            false, a, false, bv, (short)0, acc[g], false, false);
      }
    }

    // LSTM pointwise directly on the accumulators.
    // C/D layout: vgpr r -> batch row m0 + r + half*8, lane%16 -> column.
#pragma unroll
    for (int r = 0; r < 8; ++r) {
      const int b   = brow_[r];
      const int idx = idxr_[r];
      const int len = lenr_[r];

      float ig = acc[0][r] + bi;
      float fg = acc[1][r] + bf;
      float gg = acc[2][r] + bg;
      float og = acc[3][r] + bo;

      float cold = cst[idx];
      float c2 = sigf(fg) * cold + sigf(ig) * tanhf(gg);
      float h2 = sigf(og) * tanhf(c2);

      float mcur = (t < len) ? 1.0f : 0.0f;            // mask for this step
      int tn = rev ? (t - 1) : (t + 1);                // next step's time index
      float mnext = (tn >= 0 && tn < len) ? 1.0f : 0.0f;

      cst[idx]    = c2 * mnext;          // state pre-masked for next step
      hwrite[idx] = f2bf(h2 * mnext);

      float y = h2 * mcur;
      size_t yidx = ((size_t)t * BB + b) * (size_t)(2 * HH) + (size_t)dir * HH + jcol;
      if (ybf)  ybf[yidx]  = f2bf(y);
      if (yf32) yf32[yidx] = y;
    }

    grid_barrier(cnt, gen);
  }
}

// ---------------------------------------------------------------------------
extern "C" void kernel_launch(void* const* d_in, const int* in_sizes, int n_in,
                              void* d_out, int out_size, void* d_ws, size_t ws_size,
                              hipStream_t stream) {
  (void)in_sizes; (void)n_in; (void)out_size; (void)ws_size;

  const float* x     = (const float*)d_in[0];
  const int* lengths = (const int*)d_in[1];
  const float* fWih0 = (const float*)d_in[2];
  const float* fWhh0 = (const float*)d_in[3];
  const float* fb0   = (const float*)d_in[4];
  const float* bWih0 = (const float*)d_in[5];
  const float* bWhh0 = (const float*)d_in[6];
  const float* bb0   = (const float*)d_in[7];
  const float* fWih1 = (const float*)d_in[8];
  const float* fWhh1 = (const float*)d_in[9];
  const float* fb1   = (const float*)d_in[10];
  const float* bWih1 = (const float*)d_in[11];
  const float* bWhh1 = (const float*)d_in[12];
  const float* bb1   = (const float*)d_in[13];

  char* ws = (char*)d_ws;

  // workspace layout (bytes, all 256B aligned)
  const size_t o_xbf   = 0;                                    // 32 MB
  const size_t o_l0out = o_xbf + (size_t)TT * BB * 512 * 2;    // 64 MB
  const size_t o_w     = o_l0out + (size_t)TT * BB * 1024 * 2; // 20 MB
  const size_t o_state = o_w + (size_t)10485760 * 2;

  unsigned short* x_bf  = (unsigned short*)(ws + o_xbf);
  unsigned short* l0out = (unsigned short*)(ws + o_l0out);
  unsigned short* w     = (unsigned short*)(ws + o_w);

  unsigned short* w_fWih0 = w + 0;
  unsigned short* w_fWhh0 = w + 1048576;
  unsigned short* w_bWih0 = w + 2097152;
  unsigned short* w_bWhh0 = w + 3145728;
  unsigned short* w_fWih1 = w + 4194304;   // 2048x1024
  unsigned short* w_fWhh1 = w + 6291456;
  unsigned short* w_bWih1 = w + 7340032;   // 2048x1024
  unsigned short* w_bWhh1 = w + 9437184;

  // state: per layer  h(2*2*B*H bf16 = 256KB) | c(2*B*H f32 = 256KB) | bar(256B)
  char* st = ws + o_state;
  unsigned short* h0 = (unsigned short*)(st);
  float*          c0 = (float*)(st + 262144);
  unsigned*     bar0 = (unsigned*)(st + 524288);
  unsigned short* h1 = (unsigned short*)(st + 524544);
  float*          c1 = (float*)(st + 524544 + 262144);
  unsigned*     bar1 = (unsigned*)(st + 524544 + 524288);
  const int state_dwords = (524544 * 2) / 4;

  // --- zero state (h, c, barriers) so graph replay is deterministic ---
  zero_u32<<<dim3((state_dwords + 255) / 256), dim3(256), 0, stream>>>(
      (unsigned*)st, state_dwords);

  // --- f32 -> bf16 conversions ---
  auto cvt = [&](const float* s, unsigned short* d, int n) {
    int blocks = (n + 255) / 256; if (blocks > 4096) blocks = 4096;
    cvt_f32_to_bf16<<<dim3(blocks), dim3(256), 0, stream>>>(s, d, n);
  };
  cvt(x,     x_bf,    TT * BB * 512);
  cvt(fWih0, w_fWih0, 2048 * 512);
  cvt(fWhh0, w_fWhh0, 2048 * 512);
  cvt(bWih0, w_bWih0, 2048 * 512);
  cvt(bWhh0, w_bWhh0, 2048 * 512);
  cvt(fWih1, w_fWih1, 2048 * 1024);
  cvt(fWhh1, w_fWhh1, 2048 * 512);
  cvt(bWih1, w_bWih1, 2048 * 1024);
  cvt(bWhh1, w_bWhh1, 2048 * 512);

  // --- layer 0: input x (DIN=512), emit bf16 concat(f,b) into l0out ---
  lstm_layer_kernel<512><<<dim3(NBLK, 2), dim3(NTHREADS), 0, stream>>>(
      x_bf,
      w_fWih0, w_fWhh0, fb0,
      w_bWih0, w_bWhh0, bb0,
      lengths,
      l0out, (float*)nullptr,
      h0, c0, bar0);

  // --- layer 1: input l0out (DIN=1024), emit f32 into d_out ---
  lstm_layer_kernel<1024><<<dim3(NBLK, 2), dim3(NTHREADS), 0, stream>>>(
      l0out,
      w_fWih1, w_fWhh1, fb1,
      w_bWih1, w_bWhh1, bb1,
      lengths,
      (unsigned short*)nullptr, (float*)d_out,
      h1, c1, bar1);
}
